// VectorQuantizer_56581899157661
// MI455X (gfx1250) — matrix operand order repaired and checked by reading, not compiled, not run on hardware
//
#include <hip/hip_runtime.h>
#include <hip/hip_bf16.h>

// ---------------------------------------------------------------------------
// Problem constants (from reference): B=16, C=D=256, H=W=32, K=8192
//   N = B*H*W = 16384 vectors of dim 256
// Outputs (concatenated): out[4194304] f32, loss, commit, cb_loss, idx[16384]
// ---------------------------------------------------------------------------
#define VQ_N      16384
#define VQ_K      8192
#define VQ_D      256
#define VQ_OUTN   4194304   // 16*256*32*32
#define VQ_MEAN_N 4194304.0f
#define KSPLIT    4
#define TILES_PER_CHUNK (VQ_K / 16 / KSPLIT)   // 128 tiles of 16 codes
#define CODES_PER_CHUNK (VQ_K / KSPLIT)        // 2048

typedef __attribute__((ext_vector_type(16))) __bf16 v16bf;
typedef __attribute__((ext_vector_type(8)))  float  v8f;

union ABTile {
    uint4 u[2];   // two 16-byte halves
    v16bf v;      // 16 bf16 = 32 bytes
};

__device__ __forceinline__ unsigned short f2bf(float f) {
    unsigned int u = __float_as_uint(f);
    unsigned int r = u + 0x7fffu + ((u >> 16) & 1u);   // round-to-nearest-even
    return (unsigned short)(r >> 16);
}

// ---------------------------------------------------------------------------
// K1: codebook[k][d] = sum_c emb[k][c] * proj_w[d][c]   (fp32, tiny GEMM)
// ---------------------------------------------------------------------------
__global__ __launch_bounds__(256) void vq_codebook_gemm(
    const float* __restrict__ emb, const float* __restrict__ pw,
    float* __restrict__ cbf)
{
    __shared__ float sE[16][17];
    __shared__ float sW[16][17];
    const int tx = threadIdx.x & 15;
    const int ty = threadIdx.x >> 4;
    const int k0 = blockIdx.x * 16;
    const int d0 = blockIdx.y * 16;
    float acc = 0.0f;
    for (int cc = 0; cc < VQ_D; cc += 16) {
        sE[ty][tx] = emb[(k0 + ty) * VQ_D + cc + tx];
        sW[ty][tx] = pw[(d0 + ty) * VQ_D + cc + tx];
        __syncthreads();
        #pragma unroll
        for (int i = 0; i < 16; ++i)
            acc = fmaf(sE[ty][i], sW[tx][i], acc);
        __syncthreads();
    }
    cbf[(k0 + ty) * VQ_D + d0 + tx] = acc;
}

// ---------------------------------------------------------------------------
// K2: per-code bf16 convert + ||e_k||^2
// ---------------------------------------------------------------------------
__global__ __launch_bounds__(256) void vq_codebook_norm(
    const float* __restrict__ cbf, unsigned short* __restrict__ cbh,
    float* __restrict__ enorm)
{
    __shared__ float red[256];
    const int k = blockIdx.x;
    const int c = threadIdx.x;
    float v = cbf[k * VQ_D + c];
    cbh[k * VQ_D + c] = f2bf(v);
    red[c] = v * v;
    __syncthreads();
    for (int s = 128; s > 0; s >>= 1) {
        if (c < s) red[c] += red[c + s];
        __syncthreads();
    }
    if (c == 0) enorm[k] = red[0];
}

// ---------------------------------------------------------------------------
// K3: z [B,C,H,W] -> zf bf16 [N, D] (BHWC rows); zero loss accumulator
// ---------------------------------------------------------------------------
__global__ __launch_bounds__(256) void vq_z_pack(
    const float* __restrict__ z, unsigned short* __restrict__ zfh,
    float* __restrict__ accum)
{
    const int n = blockIdx.x;       // n = b*1024 + h*32 + w
    const int c = threadIdx.x;
    const int b  = n >> 10;
    const int hw = n & 1023;
    const float v = z[b * 262144 + c * 1024 + hw];
    zfh[n * VQ_D + c] = f2bf(v);
    if (n == 0 && c == 0) *accum = 0.0f;
}

// ---------------------------------------------------------------------------
// Async stage of one 16-code x 256-K bf16 tile (8 KB) into LDS.
// 256 threads x 32 bytes each, via global_load_async_to_lds_b128 (ASYNCcnt).
// ---------------------------------------------------------------------------
__device__ __forceinline__ void vq_stage_tile(
    const unsigned short* __restrict__ cbh, int ct,
    unsigned short* dst, int t)
{
    const unsigned short* src = cbh + (size_t)ct * 16 * VQ_D + t * 16;
    unsigned lds = (unsigned)(uintptr_t)(dst + t * 16);   // flat->LDS: low 32 bits
    unsigned long long ga = (unsigned long long)(uintptr_t)src;
    asm volatile(
        "global_load_async_to_lds_b128 %0, %1, off\n\t"
        "global_load_async_to_lds_b128 %0, %1, off offset:16"
        :: "v"(lds), "v"(ga) : "memory");
}

__device__ __forceinline__ void vq_wait_async() {
    asm volatile("s_wait_asynccnt 0x0" ::: "memory");
}

// ---------------------------------------------------------------------------
// K4: main distance GEMM + partial argmin.
// grid = 512 blocks (128 row-groups x 4 K-chunks), 256 threads (8 waves).
// Each wave owns 16 z-rows (A panel in 64 VGPRs) and scans 2048 codes.
// B tiles double-buffered in LDS via async loads; enorm chunk LDS-resident.
// argmin needs only (||e||^2 - 2*dot): ||z||^2 is constant per row.
// ---------------------------------------------------------------------------
__global__ __launch_bounds__(256) void vq_argmin(
    const unsigned short* __restrict__ zfh,   // [N, D] bf16
    const unsigned short* __restrict__ cbh,   // [K, D] bf16
    const float* __restrict__ enorm,          // [K]
    float* __restrict__ pbV,                  // [N * KSPLIT] partial best dist
    int*   __restrict__ pbI)                  // [N * KSPLIT] partial best idx
{
    __shared__ unsigned short shB[2][16 * VQ_D];   // 2 x 8 KB double buffer
    __shared__ float          shE[CODES_PER_CHUNK]; // 8 KB of ||e||^2

    const int t      = threadIdx.x;
    const int lane   = t & 31;
    const int wave   = t >> 5;
    const int rowblk = blockIdx.x >> 2;    // 0..127
    const int kchunk = blockIdx.x & 3;     // 0..3
    const int nbase  = rowblk * 128 + wave * 16;
    const int ct0    = kchunk * TILES_PER_CHUNK;

    // stage this chunk's ||e||^2 into LDS (8 floats per thread)
    #pragma unroll
    for (int j = 0; j < CODES_PER_CHUNK / 256; ++j)
        shE[t + j * 256] = enorm[kchunk * CODES_PER_CHUNK + t + j * 256];

    // --- A panel: 16 rows x 256 K, held in 8 v16bf chunks (64 VGPRs/lane) ---
    // ISA 16-bit A 16x32 layout: lanes 0-15 -> K offsets {0..7, 16..23},
    // lanes 16-31 -> K offsets {8..15, 24..31}; M = lane & 15.
    const int arow  = nbase + (lane & 15);
    const int akoff = (lane >> 4) * 8;
    const unsigned short* zr = zfh + (size_t)arow * VQ_D;
    ABTile A[8];
    #pragma unroll
    for (int ch = 0; ch < 8; ++ch) {
        A[ch].u[0] = *(const uint4*)(zr + ch * 32 + akoff);
        A[ch].u[1] = *(const uint4*)(zr + ch * 32 + 16 + akoff);
    }

    // --- B layout: column = lane&15; lanes 0-15 cover K {0..7, 16..23} of the chunk,
    // lanes 16-31 cover K {8..15, 24..31} (same order as A). ---
    const int bcol  = lane & 15;
    const int bkoff = (lane >> 4) * 8;

    float bestV[8];
    int   bestI[8];
    #pragma unroll
    for (int m = 0; m < 8; ++m) { bestV[m] = 3.4e38f; bestI[m] = 0; }

    vq_stage_tile(cbh, ct0, shB[0], t);    // prefill buffer 0

    for (int i = 0; i < TILES_PER_CHUNK; ++i) {
        const int cur = i & 1;
        vq_wait_async();      // this wave's async fills complete
        __syncthreads();      // all waves' fills visible; prev reads done
        if (i + 1 < TILES_PER_CHUNK)
            vq_stage_tile(cbh, ct0 + i + 1, shB[cur ^ 1], t);

        const int code = (ct0 + i) * 16 + bcol;
        const float en = shE[i * 16 + bcol];
        const unsigned short* br = &shB[cur][bcol * VQ_D + bkoff];

        v8f acc = {};
        #pragma unroll
        for (int ch = 0; ch < 8; ++ch) {
            ABTile Bt;
            Bt.u[0] = *(const uint4*)(br + ch * 32);
            Bt.u[1] = *(const uint4*)(br + ch * 32 + 16);
            acc = __builtin_amdgcn_wmma_f32_16x16x32_bf16(
                false, A[ch].v, false, Bt.v, (short)0, acc, false, false);
        }

        // D layout: lanes 0-15 hold rows 0-7 (vgpr m), lanes 16-31 rows 8-15
        #pragma unroll
        for (int m = 0; m < 8; ++m) {
            float d = en - 2.0f * acc[m];
            if (d < bestV[m]) { bestV[m] = d; bestI[m] = code; }  // codes ascend
        }
    }

    // cross-lane argmin within each 16-lane half (xor masks stay in-half)
    #pragma unroll
    for (int m = 0; m < 8; ++m) {
        float v = bestV[m];
        int   i = bestI[m];
        #pragma unroll
        for (int s = 1; s < 16; s <<= 1) {
            float vo = __shfl_xor(v, s, 32);
            int   io = __shfl_xor(i, s, 32);
            if (vo < v || (vo == v && io < i)) { v = vo; i = io; }
        }
        if ((lane & 15) == 0) {
            const int row = nbase + (lane >> 4) * 8 + m;
            pbV[row * KSPLIT + kchunk] = v;
            pbI[row * KSPLIT + kchunk] = i;
        }
    }
}

// ---------------------------------------------------------------------------
// K5: reduce the KSPLIT partial argmins per row; emit idx (int + float)
// ---------------------------------------------------------------------------
__global__ __launch_bounds__(256) void vq_argmin_reduce(
    const float* __restrict__ pbV, const int* __restrict__ pbI,
    int* __restrict__ idxw, float* __restrict__ out_idx)
{
    const int r = blockIdx.x * 256 + threadIdx.x;
    float bv = pbV[r * KSPLIT];
    int   bi = pbI[r * KSPLIT];
    #pragma unroll
    for (int j = 1; j < KSPLIT; ++j) {
        const float v = pbV[r * KSPLIT + j];
        const int   i = pbI[r * KSPLIT + j];
        if (v < bv || (v == bv && i < bi)) { bv = v; bi = i; }
    }
    idxw[r]    = bi;
    out_idx[r] = (float)bi;
}

// ---------------------------------------------------------------------------
// K6: gather codebook rows -> out (BCHW, coalesced) + MSE accumulation.
// Block owns one (b, c) plane; lanes sweep hw -> unit-stride z/out access.
// ---------------------------------------------------------------------------
__global__ __launch_bounds__(256) void vq_gather_loss(
    const float* __restrict__ z, const float* __restrict__ cbf,
    const int* __restrict__ idx, float* __restrict__ out,
    float* __restrict__ accum)
{
    __shared__ float red[256];
    const int m = blockIdx.x;        // 0..4095 : b = m>>8, c = m&255
    const int b = m >> 8;
    const int c = m & 255;
    const int t = threadIdx.x;
    float s = 0.0f;
    #pragma unroll
    for (int j = 0; j < 4; ++j) {
        const int hw   = t + j * 256;
        const int n    = b * 1024 + hw;
        const int code = idx[n];
        const int off  = m * 1024 + hw;       // == (b*256 + c)*1024 + hw
        const float zq = cbf[code * VQ_D + c];
        const float zt = z[off];
        out[off] = zq;                        // straight-through == z_quantized
        const float d = zq - zt;
        s += d * d;
    }
    red[t] = s;
    __syncthreads();
    for (int st = 128; st > 0; st >>= 1) {
        if (t < st) red[t] += red[t + st];
        __syncthreads();
    }
    if (t == 0) atomicAdd(accum, red[0]);
}

__global__ void vq_finalize(const float* __restrict__ accum,
                            float* __restrict__ loss3)
{
    const float mse = *accum / VQ_MEAN_N;
    loss3[0] = 1.25f * mse;   // loss = commitment + codebook
    loss3[1] = 0.25f * mse;   // commitment_loss
    loss3[2] = mse;           // codebook_loss
}

// ---------------------------------------------------------------------------
extern "C" void kernel_launch(void* const* d_in, const int* in_sizes, int n_in,
                              void* d_out, int out_size, void* d_ws, size_t ws_size,
                              hipStream_t stream) {
    (void)in_sizes; (void)n_in; (void)out_size; (void)ws_size;
    const float* z   = (const float*)d_in[0];   // [16,256,32,32]
    const float* emb = (const float*)d_in[1];   // [8192,256]
    const float* pw  = (const float*)d_in[2];   // [256,256]

    float* out      = (float*)d_out;            // [4194304]
    float* loss3    = out + VQ_OUTN;            // [3]
    float* out_idx  = loss3 + 3;                // [16384] as float

    // workspace carve-out
    char* ws = (char*)d_ws;
    size_t off = 0;
    float*          cbf   = (float*)(ws + off);          off += (size_t)VQ_K * VQ_D * 4;
    unsigned short* cbh   = (unsigned short*)(ws + off); off += (size_t)VQ_K * VQ_D * 2;
    unsigned short* zfh   = (unsigned short*)(ws + off); off += (size_t)VQ_N * VQ_D * 2;
    float*          enorm = (float*)(ws + off);          off += (size_t)VQ_K * 4;
    int*            idxw  = (int*)(ws + off);            off += (size_t)VQ_N * 4;
    float*          pbV   = (float*)(ws + off);          off += (size_t)VQ_N * KSPLIT * 4;
    int*            pbI   = (int*)(ws + off);            off += (size_t)VQ_N * KSPLIT * 4;
    float*          accum = (float*)(ws + off);          off += 256;

    vq_codebook_gemm<<<dim3(VQ_K / 16, VQ_D / 16), 256, 0, stream>>>(emb, pw, cbf);
    vq_codebook_norm<<<VQ_K, 256, 0, stream>>>(cbf, cbh, enorm);
    vq_z_pack<<<VQ_N, 256, 0, stream>>>(z, zfh, accum);
    vq_argmin<<<(VQ_N / 128) * KSPLIT, 256, 0, stream>>>(zfh, cbh, enorm, pbV, pbI);
    vq_argmin_reduce<<<VQ_N / 256, 256, 0, stream>>>(pbV, pbI, idxw, out_idx);
    vq_gather_loss<<<4096, 256, 0, stream>>>(z, cbf, idxw, out, accum);
    vq_finalize<<<1, 1, 0, stream>>>(accum, loss3);
}
